// BERTEncoderBlock_33088428049020
// MI455X (gfx1250) — compile-verified, matrix operand order checked by reading, not run on hardware
//
#include <hip/hip_runtime.h>
#include <hip/hip_bf16.h>

// ---------------------------------------------------------------------------
// BERT encoder block for MI455X (gfx1250), bf16 WMMA GEMM core with
// async global->LDS double-buffered tile staging (GLOBAL_LOAD_ASYNC_TO_LDS).
// ---------------------------------------------------------------------------

typedef __attribute__((ext_vector_type(16))) __bf16 bfrag_t; // 16 bf16 = 8 VGPRs
typedef __attribute__((ext_vector_type(8)))  float  accf_t;  // 16x16 f32 C/D tile

#if __has_builtin(__builtin_amdgcn_global_load_async_to_lds_b128)
#define HAVE_ASYNC_COPY 1
typedef __attribute__((ext_vector_type(4))) int v4i_t;
typedef __attribute__((address_space(1))) v4i_t gv4i_t; // gv4i_t* = v4i AS(1)*
typedef __attribute__((address_space(3))) v4i_t lv4i_t; // lv4i_t* = v4i AS(3)*
__device__ __forceinline__ void wait_async0() {
#if __has_builtin(__builtin_amdgcn_s_wait_asynccnt)
  __builtin_amdgcn_s_wait_asynccnt(0);
#else
  asm volatile("s_wait_asynccnt 0" ::: "memory");
#endif
}
#endif

#define EMBED 1024
#define FFDIM 4096
#define SEQ   2048
#define BATCH 8
#define NTOK  (SEQ * BATCH)

#define BM 128
#define BN 128
#define BK 32

__device__ __forceinline__ unsigned short f2bf(float f) {
  unsigned int u = __float_as_uint(f);
  return (unsigned short)((u + 0x7FFFu + ((u >> 16) & 1u)) >> 16); // RNE
}

// Load one 16x32 bf16 fragment from an LDS tile stored row-major, 32 bf16/row.
// CDNA5 16-bit layout: VGPR g holds K pair k0 = (g<4 ? 2g : 8+2g) + 8*half.
__device__ __forceinline__ bfrag_t frag_ld(const unsigned int* smem_u32,
                                           int row_u32, int half) {
  union { bfrag_t v; unsigned int u[8]; } f;
#pragma unroll
  for (int g = 0; g < 8; ++g) {
    int k0 = ((g < 4) ? (2 * g) : (8 + 2 * g)) + 8 * half;
    f.u[g] = smem_u32[row_u32 + (k0 >> 1)];
  }
  return f.v;
}

// C[m,n] = act(scale * sum_k A[m,k]*Bt[n,k] + bias[n]), A/Bt bf16, C f32/bf16.
// Block: 256 threads (8 waves). Tile 128x128x32. Wave patch 64x32 (4x2 WMMAs).
__global__ void __launch_bounds__(256)
gemm_bf16(const unsigned short* __restrict__ A,
          const unsigned short* __restrict__ Bt,
          const float* __restrict__ bias,
          float* __restrict__ Cf, unsigned short* __restrict__ Cb,
          int M, int N, int K, float scale, int relu,
          long long sA, long long sB, long long sC) {
#ifdef HAVE_ASYNC_COPY
  __shared__ uint4 As4[2][512]; // 2 x 8 KB, double-buffered
  __shared__ uint4 Bs4[2][512];
#else
  __shared__ uint4 As4[1][512];
  __shared__ uint4 Bs4[1][512];
#endif

  const int tid  = threadIdx.x;
  const int lane = tid & 31;
  const int wv   = tid >> 5;
  const int wm   = (wv >> 2) * 64; // wave M offset in tile
  const int wn   = (wv & 3) * 32;  // wave N offset in tile
  const int half = lane >> 4;
  const int lr   = lane & 15;

  const long long batch = blockIdx.z;
  A  += batch * sA;
  Bt += batch * sB;
  const long long cOff = batch * sC;

  const int m0 = blockIdx.y * BM;
  const int n0 = blockIdx.x * BN;
  const int Ku  = K >> 1;  // bf16 pairs per row
  const int rs4 = Ku >> 2; // uint4 per row

  const uint4* Ag4 = reinterpret_cast<const uint4*>(A) + (long long)m0 * rs4;
  const uint4* Bg4 = reinterpret_cast<const uint4*>(Bt) + (long long)n0 * rs4;

  accf_t acc[4][2];
#pragma unroll
  for (int i = 0; i < 4; ++i)
#pragma unroll
    for (int j = 0; j < 2; ++j)
#pragma unroll
      for (int g = 0; g < 8; ++g) acc[i][j][g] = 0.0f;

  auto compute_step = [&](int b) {
    const unsigned int* Asu = reinterpret_cast<const unsigned int*>(&As4[b][0]);
    const unsigned int* Bsu = reinterpret_cast<const unsigned int*>(&Bs4[b][0]);
    bfrag_t af[4], bfr[2];
#pragma unroll
    for (int ms = 0; ms < 4; ++ms)
      af[ms] = frag_ld(Asu, (wm + ms * 16 + lr) * 16, half);
#pragma unroll
    for (int ns = 0; ns < 2; ++ns)
      bfr[ns] = frag_ld(Bsu, (wn + ns * 16 + lr) * 16, half);
#pragma unroll
    for (int ms = 0; ms < 4; ++ms)
#pragma unroll
      for (int ns = 0; ns < 2; ++ns)
        acc[ms][ns] = __builtin_amdgcn_wmma_f32_16x16x32_bf16(
            false, af[ms], false, bfr[ns], (short)0, acc[ms][ns], false, false);
  };

  const int nK = K / BK;

#ifdef HAVE_ASYNC_COPY
  // Async double-buffered pipeline: copy(kt+1) overlaps WMMA on tile kt.
  auto stage = [&](int kt, int b) {
#pragma unroll
    for (int i = 0; i < 2; ++i) {
      int j = tid + i * 256; // 512 x 16B chunks per tile
      int r = j >> 2, c = j & 3;
      const uint4* ga = Ag4 + (long long)r * rs4 + kt * 4 + c;
      const uint4* gb = Bg4 + (long long)r * rs4 + kt * 4 + c;
      __builtin_amdgcn_global_load_async_to_lds_b128(
          (gv4i_t*)ga, (lv4i_t*)&As4[b][j], 0, 0);
      __builtin_amdgcn_global_load_async_to_lds_b128(
          (gv4i_t*)gb, (lv4i_t*)&Bs4[b][j], 0, 0);
    }
  };

  stage(0, 0);
  for (int kt = 0; kt < nK; ++kt) {
    wait_async0();     // this wave's pending copy (tile kt) has landed in LDS
    __syncthreads();   // all waves' copies landed; prev buffer free to reuse
    if (kt + 1 < nK) stage(kt + 1, (kt + 1) & 1);
    compute_step(kt & 1);
  }
#else
  // Fallback: vectorized synchronous staging (b128 loads + b128 LDS stores).
  for (int kt = 0; kt < nK; ++kt) {
    __syncthreads();
#pragma unroll
    for (int i = 0; i < 2; ++i) {
      int j = tid + i * 256;
      int r = j >> 2, c = j & 3;
      As4[0][j] = Ag4[(long long)r * rs4 + kt * 4 + c];
      Bs4[0][j] = Bg4[(long long)r * rs4 + kt * 4 + c];
    }
    if (tid == 0 && kt + 1 < nK) {
      __builtin_prefetch(Ag4 + (kt + 1) * 4, 0, 1); // global_prefetch_b8
      __builtin_prefetch(Bg4 + (kt + 1) * 4, 0, 1);
    }
    __syncthreads();
    compute_step(0);
  }
#endif

  // Epilogue: C layout VGPR g -> M = g + 8*half, N = lane&15.
#pragma unroll
  for (int ms = 0; ms < 4; ++ms) {
#pragma unroll
    for (int ns = 0; ns < 2; ++ns) {
      const int ccol = n0 + wn + ns * 16 + lr;
      const float bv = bias ? bias[ccol] : 0.0f;
#pragma unroll
      for (int g = 0; g < 8; ++g) {
        const int rrow = m0 + wm + ms * 16 + half * 8 + g;
        float v = acc[ms][ns][g] * scale + bv;
        if (relu) v = fmaxf(v, 0.0f);
        const long long idx = cOff + (long long)rrow * N + ccol;
        if (Cf) Cf[idx] = v;
        if (Cb) Cb[idx] = f2bf(v);
      }
    }
  }
}

// ---------------------------------------------------------------------------
// Elementwise fp32 -> bf16
__global__ void __launch_bounds__(256)
cvt_f32_bf16(const float* __restrict__ in, unsigned short* __restrict__ out,
             long long n) {
  long long i = (long long)blockIdx.x * 256 + threadIdx.x;
  if (i < n) out[i] = f2bf(in[i]);
}

// fp32 [rows][cols] -> bf16 [cols][rows] (weight pre-transpose). Block (32,8).
__global__ void __launch_bounds__(256)
transpose_cvt_f32_bf16(const float* __restrict__ in,
                       unsigned short* __restrict__ out, int rows, int cols) {
  __shared__ float t[32][33];
  const int c0 = blockIdx.x * 32;
  const int r0 = blockIdx.y * 32;
  for (int i = threadIdx.y; i < 32; i += 8)
    t[i][threadIdx.x] = in[(long long)(r0 + i) * cols + (c0 + threadIdx.x)];
  __syncthreads();
  for (int i = threadIdx.y; i < 32; i += 8)
    out[(long long)(c0 + i) * rows + (r0 + threadIdx.x)] = f2bf(t[threadIdx.x][i]);
}

// bf16 [rows][cols] -> bf16 [cols][rows], batched via blockIdx.z. Block (32,8).
__global__ void __launch_bounds__(256)
transpose_bf16(const unsigned short* __restrict__ in,
               unsigned short* __restrict__ out, int rows, int cols,
               long long sIn, long long sOut) {
  __shared__ unsigned short t[32][33];
  in  += (long long)blockIdx.z * sIn;
  out += (long long)blockIdx.z * sOut;
  const int c0 = blockIdx.x * 32;
  const int r0 = blockIdx.y * 32;
  for (int i = threadIdx.y; i < 32; i += 8)
    t[i][threadIdx.x] = in[(long long)(r0 + i) * cols + (c0 + threadIdx.x)];
  __syncthreads();
  for (int i = threadIdx.y; i < 32; i += 8)
    out[(long long)(c0 + i) * rows + (r0 + threadIdx.x)] = t[threadIdx.x][i];
}

// Row softmax over 2048 fp32 elements -> bf16. One block per row, 8 elem/thread.
__global__ void __launch_bounds__(256)
softmax_rows(const float* __restrict__ S, unsigned short* __restrict__ P) {
  __shared__ float red[256];
  const int tid = threadIdx.x;
  const long long base = (long long)blockIdx.x * 2048;
  float v[8];
  float mx = -3.402823e38f;
#pragma unroll
  for (int i = 0; i < 8; ++i) {
    v[i] = S[base + tid + i * 256];
    mx = fmaxf(mx, v[i]);
  }
  red[tid] = mx;
  __syncthreads();
  for (int s = 128; s > 0; s >>= 1) {
    if (tid < s) red[tid] = fmaxf(red[tid], red[tid + s]);
    __syncthreads();
  }
  mx = red[0];
  __syncthreads();
  float sum = 0.0f;
#pragma unroll
  for (int i = 0; i < 8; ++i) {
    v[i] = __expf(v[i] - mx);
    sum += v[i];
  }
  red[tid] = sum;
  __syncthreads();
  for (int s = 128; s > 0; s >>= 1) {
    if (tid < s) red[tid] += red[tid + s];
    __syncthreads();
  }
  const float inv = 1.0f / red[0];
#pragma unroll
  for (int i = 0; i < 8; ++i) P[base + tid + i * 256] = f2bf(v[i] * inv);
}

// out = LN(a + b) * gamma + beta over E=1024; optional bf16 copy. Block=256.
__global__ void __launch_bounds__(256)
layernorm_residual(const float* __restrict__ a, const float* __restrict__ b,
                   const float* __restrict__ gamma, const float* __restrict__ beta,
                   float* __restrict__ out, unsigned short* __restrict__ outb) {
  __shared__ float r1[256];
  __shared__ float r2[256];
  const int tid = threadIdx.x;
  const long long base = (long long)blockIdx.x * EMBED;
  float v[4];
  float s = 0.0f, s2 = 0.0f;
#pragma unroll
  for (int i = 0; i < 4; ++i) {
    const float t = a[base + tid + i * 256] + b[base + tid + i * 256];
    v[i] = t;
    s += t;
    s2 += t * t;
  }
  r1[tid] = s;
  r2[tid] = s2;
  __syncthreads();
  for (int st = 128; st > 0; st >>= 1) {
    if (tid < st) { r1[tid] += r1[tid + st]; r2[tid] += r2[tid + st]; }
    __syncthreads();
  }
  const float mu   = r1[0] * (1.0f / EMBED);
  const float var  = r2[0] * (1.0f / EMBED) - mu * mu;
  const float rstd = rsqrtf(var + 1e-5f);
#pragma unroll
  for (int i = 0; i < 4; ++i) {
    const int c = tid + i * 256;
    const float y = (v[i] - mu) * rstd * gamma[c] + beta[c];
    out[base + c] = y;
    if (outb) outb[base + c] = f2bf(y);
  }
}

// ---------------------------------------------------------------------------
extern "C" void kernel_launch(void* const* d_in, const int* in_sizes, int n_in,
                              void* d_out, int out_size, void* d_ws, size_t ws_size,
                              hipStream_t stream) {
  (void)in_sizes; (void)n_in; (void)out_size; (void)ws_size;

  const float* x   = (const float*)d_in[0];
  const float* Wq  = (const float*)d_in[1];
  const float* bq  = (const float*)d_in[2];
  const float* Wk  = (const float*)d_in[3];
  const float* bk  = (const float*)d_in[4];
  const float* Wv  = (const float*)d_in[5];
  const float* bv  = (const float*)d_in[6];
  const float* Wo  = (const float*)d_in[7];
  const float* bo  = (const float*)d_in[8];
  const float* g1  = (const float*)d_in[9];
  const float* b1  = (const float*)d_in[10];
  const float* g2  = (const float*)d_in[11];
  const float* b2  = (const float*)d_in[12];
  const float* W1  = (const float*)d_in[13];
  const float* bf1 = (const float*)d_in[14];
  const float* W2  = (const float*)d_in[15];
  const float* bf2 = (const float*)d_in[16];
  float* out = (float*)d_out;

  char* base = (char*)d_ws;
  size_t off = 0;
  auto alloc = [&](size_t bytes) -> void* {
    void* r = base + off;
    off += (bytes + 255) & ~(size_t)255;
    return r;
  };

  const size_t tokE2 = (size_t)NTOK * EMBED * 2;  // bf16 [NTOK,E]
  const size_t tokE4 = (size_t)NTOK * EMBED * 4;  // f32  [NTOK,E]

  unsigned short* Wtq  = (unsigned short*)alloc((size_t)EMBED * EMBED * 2);
  unsigned short* Wtk  = (unsigned short*)alloc((size_t)EMBED * EMBED * 2);
  unsigned short* Wtv  = (unsigned short*)alloc((size_t)EMBED * EMBED * 2);
  unsigned short* Wto  = (unsigned short*)alloc((size_t)EMBED * EMBED * 2);
  unsigned short* Wt1  = (unsigned short*)alloc((size_t)FFDIM * EMBED * 2);
  unsigned short* Wt2  = (unsigned short*)alloc((size_t)EMBED * FFDIM * 2);
  unsigned short* xb   = (unsigned short*)alloc(tokE2);
  unsigned short* Qb   = (unsigned short*)alloc(tokE2);
  unsigned short* Kb   = (unsigned short*)alloc(tokE2);
  unsigned short* Vb   = (unsigned short*)alloc(tokE2);
  unsigned short* Vtb  = (unsigned short*)alloc(tokE2);                       // [B][E][S]
  float*          scor = (float*)alloc((size_t)BATCH * SEQ * SEQ * 4);        // [B][S][S]
  unsigned short* P    = (unsigned short*)alloc((size_t)BATCH * SEQ * SEQ * 2);
  unsigned short* ctx  = (unsigned short*)alloc(tokE2);
  float*          attp = (float*)alloc(tokE4);
  float*          h    = (float*)alloc(tokE4);
  unsigned short* hb   = (unsigned short*)alloc(tokE2);
  unsigned short* ff1  = (unsigned short*)alloc((size_t)NTOK * FFDIM * 2);
  float*          ff2  = (float*)alloc(tokE4);

  const dim3 blk(256);
  const dim3 tb(32, 8);

  // ---- precision conversion -------------------------------------------------
  {
    long long n = (long long)NTOK * EMBED;
    cvt_f32_bf16<<<dim3((unsigned)(n / 256)), blk, 0, stream>>>(x, xb, n);
  }
  transpose_cvt_f32_bf16<<<dim3(EMBED / 32, EMBED / 32), tb, 0, stream>>>(Wq, Wtq, EMBED, EMBED);
  transpose_cvt_f32_bf16<<<dim3(EMBED / 32, EMBED / 32), tb, 0, stream>>>(Wk, Wtk, EMBED, EMBED);
  transpose_cvt_f32_bf16<<<dim3(EMBED / 32, EMBED / 32), tb, 0, stream>>>(Wv, Wtv, EMBED, EMBED);
  transpose_cvt_f32_bf16<<<dim3(EMBED / 32, EMBED / 32), tb, 0, stream>>>(Wo, Wto, EMBED, EMBED);
  transpose_cvt_f32_bf16<<<dim3(FFDIM / 32, EMBED / 32), tb, 0, stream>>>(W1, Wt1, EMBED, FFDIM);
  transpose_cvt_f32_bf16<<<dim3(EMBED / 32, FFDIM / 32), tb, 0, stream>>>(W2, Wt2, FFDIM, EMBED);

  // ---- Q/K/V projections ----------------------------------------------------
  gemm_bf16<<<dim3(EMBED / BN, NTOK / BM, 1), blk, 0, stream>>>(
      xb, Wtq, bq, nullptr, Qb, NTOK, EMBED, EMBED, 1.0f, 0, 0, 0, 0);
  gemm_bf16<<<dim3(EMBED / BN, NTOK / BM, 1), blk, 0, stream>>>(
      xb, Wtk, bk, nullptr, Kb, NTOK, EMBED, EMBED, 1.0f, 0, 0, 0, 0);
  gemm_bf16<<<dim3(EMBED / BN, NTOK / BM, 1), blk, 0, stream>>>(
      xb, Wtv, bv, nullptr, Vb, NTOK, EMBED, EMBED, 1.0f, 0, 0, 0, 0);

  // V^T per batch for P*V (B operand must be [N,K] row-major)
  transpose_bf16<<<dim3(EMBED / 32, SEQ / 32, BATCH), tb, 0, stream>>>(
      Vb, Vtb, SEQ, EMBED, (long long)SEQ * EMBED, (long long)EMBED * SEQ);

  // ---- attention ------------------------------------------------------------
  // scores = (Q K^T) / sqrt(E); K itself is the [N,K] operand here.
  gemm_bf16<<<dim3(SEQ / BN, SEQ / BM, BATCH), blk, 0, stream>>>(
      Qb, Kb, nullptr, scor, nullptr, SEQ, SEQ, EMBED, 0.03125f, 0,
      (long long)SEQ * EMBED, (long long)SEQ * EMBED, (long long)SEQ * SEQ);

  softmax_rows<<<dim3(BATCH * SEQ), blk, 0, stream>>>(scor, P);

  // ctx = P V
  gemm_bf16<<<dim3(EMBED / BN, SEQ / BM, BATCH), blk, 0, stream>>>(
      P, Vtb, nullptr, nullptr, ctx, SEQ, EMBED, SEQ, 1.0f, 0,
      (long long)SEQ * SEQ, (long long)EMBED * SEQ, (long long)SEQ * EMBED);

  // attn output projection
  gemm_bf16<<<dim3(EMBED / BN, NTOK / BM, 1), blk, 0, stream>>>(
      ctx, Wto, bo, attp, nullptr, NTOK, EMBED, EMBED, 1.0f, 0, 0, 0, 0);

  // h = LN(x + attn)
  layernorm_residual<<<dim3(NTOK), blk, 0, stream>>>(x, attp, g1, b1, h, hb);

  // ---- FFN ------------------------------------------------------------------
  gemm_bf16<<<dim3(FFDIM / BN, NTOK / BM, 1), blk, 0, stream>>>(
      hb, Wt1, bf1, nullptr, ff1, NTOK, FFDIM, EMBED, 1.0f, 1, 0, 0, 0);
  gemm_bf16<<<dim3(EMBED / BN, NTOK / BM, 1), blk, 0, stream>>>(
      ff1, Wt2, bf2, ff2, nullptr, NTOK, EMBED, FFDIM, 1.0f, 0, 0, 0, 0);

  // out = LN(h + ffn)
  layernorm_residual<<<dim3(NTOK), blk, 0, stream>>>(h, ff2, g2, b2, out, nullptr);
}
